// E8RHTLinear_66795331387476
// MI455X (gfx1250) — matrix operand order repaired and checked by reading, not compiled, not run on hardware
//
#include <hip/hip_runtime.h>
#include <hip/hip_bf16.h>

typedef __attribute__((ext_vector_type(16))) __bf16 v16bf;
typedef __attribute__((ext_vector_type(8)))  __bf16 v8bf;
typedef __attribute__((ext_vector_type(8)))  float  v8f;

// Problem sizes (fixed by the reference)
#define KDIM   4096   // IN_F == N_PAD
#define NOUT   4096   // M_PAD == OUT_F
#define MROWS  8192   // 4 * 2048

#if __has_builtin(__builtin_amdgcn_global_load_async_to_lds_b128) && \
    __has_builtin(__builtin_amdgcn_s_wait_asynccnt)
#define USE_ASYNC_COPY 1
#else
#define USE_ASYNC_COPY 0
#endif

// GCC-style 4x i32 vector (matches the builtin's parameter type)
typedef int v4i_t __attribute__((vector_size(16)));
typedef __attribute__((address_space(1))) v4i_t as1_v4i;
typedef __attribute__((address_space(3))) v4i_t as3_v4i;

// Generic->global pointer: identity value (ISA 10.2)
__device__ __forceinline__ as1_v4i* gbl_cast(const void* p) {
    return (as1_v4i*)(unsigned long long)p;
}
// Generic->LDS pointer: ISA 10.2 aperture rule: LDS address is the low 32 bits of addr
__device__ __forceinline__ as3_v4i* lds_cast(const void* p) {
    return (as3_v4i*)(unsigned int)(unsigned long long)p;
}

// ---------------------------------------------------------------------------
// In-register 16-point Hadamard (bit factors commute -> any stage order)
// ---------------------------------------------------------------------------
__device__ __forceinline__ void fht16(float v[16]) {
#pragma unroll
    for (int h = 1; h < 16; h <<= 1) {
#pragma unroll
        for (int p = 0; p < 8; ++p) {
            const int j = p & (h - 1);
            const int a = ((p - j) << 1) | j;
            const float s0 = v[a];
            const float s1 = v[a + h];
            v[a]     = s0 + s1;
            v[a + h] = s0 - s1;
        }
    }
}

// ---------------------------------------------------------------------------
// Kernel 1: dequantize W = codebook[Qidxs] -> bf16 hi/lo split, K-contiguous
// ---------------------------------------------------------------------------
__global__ void __launch_bounds__(256)
dequant_kernel(const int* __restrict__ Qidxs, const float* __restrict__ codebook,
               __bf16* __restrict__ whi, __bf16* __restrict__ wlo) {
    const int e = blockIdx.x * 256 + threadIdx.x;   // 0 .. 4096*512-1
    const int idx = Qidxs[e];
    const float* cb = codebook + idx * 8;
    const size_t base = (size_t)e * 8;              // == m*4096 + c8*8
#pragma unroll
    for (int j = 0; j < 8; ++j) {
        const float w = cb[j];
        const __bf16 h = (__bf16)w;
        const __bf16 l = (__bf16)(w - (float)h);
        whi[base + j] = h;
        wlo[base + j] = l;
    }
}

// ---------------------------------------------------------------------------
// FHT-4096 as H16 (x) H16 (x) H16: three register phases, two LDS exchanges.
// LDS layout: logical index j stored at phys(j) = (j>>4)*20 + (j&15)
// ---------------------------------------------------------------------------

// Kernel 2: x_rht = FHT(SV * x) * (1/64) -> bf16 hi/lo split
__global__ void __launch_bounds__(256)
fht_in_kernel(const float* __restrict__ x, const float* __restrict__ SV,
              __bf16* __restrict__ xhi, __bf16* __restrict__ xlo) {
    __shared__ __align__(16) float lbuf[256 * 20];
    const int row = blockIdx.x;
    const int t   = threadIdx.x;
    const float* xr = x + ((size_t)row << 12);

    float v[16];

    // ---- Phase A: contiguous 16 elems (index bits 0-3) ----
    {
        const float4* xr4 = (const float4*)(xr + t * 16);
        const float4* sv4 = (const float4*)(SV + t * 16);
#pragma unroll
        for (int q = 0; q < 4; ++q) {
            const float4 a = xr4[q];
            const float4 s = sv4[q];
            v[q * 4 + 0] = a.x * s.x;
            v[q * 4 + 1] = a.y * s.y;
            v[q * 4 + 2] = a.z * s.z;
            v[q * 4 + 3] = a.w * s.w;
        }
        fht16(v);
        float4* dst = (float4*)(lbuf + t * 20);
#pragma unroll
        for (int q = 0; q < 4; ++q)
            dst[q] = make_float4(v[q * 4 + 0], v[q * 4 + 1], v[q * 4 + 2], v[q * 4 + 3]);
    }
    __syncthreads();

    // ---- Phase B: index bits 4-7 (thread-private locations) ----
    {
        const int tb = (t & 15) + 320 * (t >> 4);
#pragma unroll
        for (int i = 0; i < 16; ++i) v[i] = lbuf[tb + 20 * i];
        fht16(v);
#pragma unroll
        for (int i = 0; i < 16; ++i) lbuf[tb + 20 * i] = v[i];
    }
    __syncthreads();

    // ---- Phase C: index bits 8-11, then scale + bf16 hi/lo split + store ----
    {
        const int tc = (t & 15) + 20 * (t >> 4);
#pragma unroll
        for (int i = 0; i < 16; ++i) v[i] = lbuf[tc + 320 * i];
        fht16(v);
        const float s = 1.0f / 64.0f;   // 1/sqrt(4096)
        const size_t base = (size_t)row << 12;
#pragma unroll
        for (int i = 0; i < 16; ++i) {
            const float val = v[i] * s;
            const __bf16 h = (__bf16)val;
            const __bf16 l = (__bf16)(val - (float)h);
            xhi[base + t + 256 * i] = h;   // j = t + 256*i (coalesced)
            xlo[base + t + 256 * i] = l;
        }
    }
}

// Kernel 4: y = FHT(y_rht) * (1/64) * Wscale * SU   (in-place on d_out)
__global__ void __launch_bounds__(256)
fht_out_kernel(float* __restrict__ y, const float* __restrict__ SU,
               const float* __restrict__ Wscale) {
    __shared__ __align__(16) float lbuf[256 * 20];
    const int row = blockIdx.x;
    const int t   = threadIdx.x;
    float* yr = y + ((size_t)row << 12);

    float v[16];
    {
        const float4* yr4 = (const float4*)(yr + t * 16);
#pragma unroll
        for (int q = 0; q < 4; ++q) {
            const float4 a = yr4[q];
            v[q * 4 + 0] = a.x; v[q * 4 + 1] = a.y;
            v[q * 4 + 2] = a.z; v[q * 4 + 3] = a.w;
        }
        fht16(v);
        float4* dst = (float4*)(lbuf + t * 20);
#pragma unroll
        for (int q = 0; q < 4; ++q)
            dst[q] = make_float4(v[q * 4 + 0], v[q * 4 + 1], v[q * 4 + 2], v[q * 4 + 3]);
    }
    __syncthreads();
    {
        const int tb = (t & 15) + 320 * (t >> 4);
#pragma unroll
        for (int i = 0; i < 16; ++i) v[i] = lbuf[tb + 20 * i];
        fht16(v);
#pragma unroll
        for (int i = 0; i < 16; ++i) lbuf[tb + 20 * i] = v[i];
    }
    __syncthreads();
    {
        const int tc = (t & 15) + 20 * (t >> 4);
#pragma unroll
        for (int i = 0; i < 16; ++i) v[i] = lbuf[tc + 320 * i];
        fht16(v);
        const float s = (1.0f / 64.0f) * Wscale[0];
#pragma unroll
        for (int i = 0; i < 16; ++i) {
            const int j = t + 256 * i;
            yr[j] = v[i] * s * SU[j];
        }
    }
}

// ---------------------------------------------------------------------------
// Kernel 3: y_rht = x_rht @ W^T  (split-bf16, fp32 accumulate, WMMA 16x16x32)
// Block tile 128x128, 8 waves (wave32), each wave 32x64 (2x4 C fragments).
// ---------------------------------------------------------------------------
__device__ __forceinline__ v16bf pack16(const __bf16* p0, const __bf16* p1) {
    const v8bf a = *(const v8bf*)p0;
    const v8bf b = *(const v8bf*)p1;
    v16bf r;
#pragma unroll
    for (int i = 0; i < 8; ++i) { r[i] = a[i]; r[i + 8] = b[i]; }
    return r;
}

#if USE_ASYNC_COPY
// Async double-buffered variant: GLOBAL_LOAD_ASYNC_TO_LDS_B128 writes straight
// to LDS (no VGPR staging), one barrier per K-chunk, next chunk's loads overlap
// the current chunk's WMMAs.
__global__ void __launch_bounds__(256, 1)
gemm_kernel(const __bf16* __restrict__ XHI, const __bf16* __restrict__ XLO,
            const __bf16* __restrict__ WHI, const __bf16* __restrict__ WLO,
            float* __restrict__ Y) {
    constexpr int KC  = 32;   // K chunk staged in LDS
    constexpr int LDR = 40;   // padded row (bf16): 80B stride -> conflict-free

    __shared__ __align__(16) __bf16 sAh[2][128 * LDR];
    __shared__ __align__(16) __bf16 sAl[2][128 * LDR];
    __shared__ __align__(16) __bf16 sBh[2][128 * LDR];
    __shared__ __align__(16) __bf16 sBl[2][128 * LDR];

    const int tid  = threadIdx.x;
    const int lane = tid & 31;
    const int wid  = tid >> 5;
    const int wm   = wid & 3;    // wave M offset = wm*32
    const int wn   = wid >> 2;   // wave N offset = wn*64
    const int mBase = blockIdx.y * 128;
    const int nBase = blockIdx.x * 128;

    const v8f vzero = {0.f, 0.f, 0.f, 0.f, 0.f, 0.f, 0.f, 0.f};
    v8f acc[2][4];
#pragma unroll
    for (int i = 0; i < 2; ++i)
#pragma unroll
        for (int j = 0; j < 4; ++j) acc[i][j] = vzero;

    const int fm    = lane & 15;
    const int khalf = lane >> 4;
    const int crow  = tid >> 2;        // copy row within 64-row rep
    const int cc8   = (tid & 3) * 8;   // copy 16B chunk within row

#define ISSUE_CHUNK(K0, PAR)                                                   \
    do {                                                                       \
        _Pragma("unroll")                                                      \
        for (int rep = 0; rep < 2; ++rep) {                                    \
            const int row = rep * 64 + crow;                                   \
            const size_t gA = ((size_t)(mBase + row) << 12) + (K0) + cc8;      \
            const size_t gB = ((size_t)(nBase + row) << 12) + (K0) + cc8;      \
            const int lOff = row * LDR + cc8;                                  \
            __builtin_amdgcn_global_load_async_to_lds_b128(                    \
                gbl_cast(XHI + gA), lds_cast(&sAh[PAR][lOff]), 0, 0);          \
            __builtin_amdgcn_global_load_async_to_lds_b128(                    \
                gbl_cast(XLO + gA), lds_cast(&sAl[PAR][lOff]), 0, 0);          \
            __builtin_amdgcn_global_load_async_to_lds_b128(                    \
                gbl_cast(WHI + gB), lds_cast(&sBh[PAR][lOff]), 0, 0);          \
            __builtin_amdgcn_global_load_async_to_lds_b128(                    \
                gbl_cast(WLO + gB), lds_cast(&sBl[PAR][lOff]), 0, 0);          \
        }                                                                      \
    } while (0)

    ISSUE_CHUNK(0, 0);

    constexpr int NCHUNK = KDIM / KC;
    for (int it = 0; it < NCHUNK; ++it) {
        const int par = it & 1;
        // own-wave async copies for chunk `it` complete, then make all waves'
        // copies visible
        __builtin_amdgcn_s_wait_asynccnt(0);
        __syncthreads();

        // overlap next chunk's global->LDS under the WMMAs (other buffer; its
        // previous readers finished before the barrier above)
        if (it + 1 < NCHUNK) ISSUE_CHUNK((it + 1) * KC, par ^ 1);

        // ---- load fragments (ISA 7.12.2 layouts) ----
        v16bf ah[2], al[2], bh[4], bl[4];
#pragma unroll
        for (int mi = 0; mi < 2; ++mi) {
            const int m = wm * 32 + mi * 16 + fm;
            const __bf16* p = &sAh[par][m * LDR + khalf * 8];   // K runs: k, k+16
            const __bf16* q = &sAl[par][m * LDR + khalf * 8];
            ah[mi] = pack16(p, p + 16);
            al[mi] = pack16(q, q + 16);
        }
#pragma unroll
        for (int ni = 0; ni < 4; ++ni) {
            const int n = wn * 64 + ni * 16 + fm;
            const __bf16* p = &sBh[par][n * LDR + khalf * 8];   // K runs: k, k+16
            const __bf16* q = &sBl[par][n * LDR + khalf * 8];
            bh[ni] = pack16(p, p + 16);
            bl[ni] = pack16(q, q + 16);
        }

        // ---- 3-term split-bf16 WMMA accumulate ----
#pragma unroll
        for (int mi = 0; mi < 2; ++mi)
#pragma unroll
            for (int ni = 0; ni < 4; ++ni) {
                acc[mi][ni] = __builtin_amdgcn_wmma_f32_16x16x32_bf16(
                    false, ah[mi], false, bh[ni], (short)0, acc[mi][ni], false, false);
                acc[mi][ni] = __builtin_amdgcn_wmma_f32_16x16x32_bf16(
                    false, ah[mi], false, bl[ni], (short)0, acc[mi][ni], false, false);
                acc[mi][ni] = __builtin_amdgcn_wmma_f32_16x16x32_bf16(
                    false, al[mi], false, bh[ni], (short)0, acc[mi][ni], false, false);
            }
    }
#undef ISSUE_CHUNK

    // ---- epilogue: C/D layout -> global (m = r + 8*(lane/16), n = lane%16) ----
#pragma unroll
    for (int mi = 0; mi < 2; ++mi) {
#pragma unroll
        for (int ni = 0; ni < 4; ++ni) {
            const int m0 = mBase + wm * 32 + mi * 16 + khalf * 8;
            const int n  = nBase + wn * 64 + ni * 16 + fm;
#pragma unroll
            for (int r = 0; r < 8; ++r)
                Y[((size_t)(m0 + r) << 12) + n] = acc[mi][ni][r];
        }
    }
}

#else  // !USE_ASYNC_COPY: register-prefetch fallback (spill-free via launch bounds)

__global__ void __launch_bounds__(256, 1)
gemm_kernel(const __bf16* __restrict__ XHI, const __bf16* __restrict__ XLO,
            const __bf16* __restrict__ WHI, const __bf16* __restrict__ WLO,
            float* __restrict__ Y) {
    constexpr int KC  = 32;
    constexpr int LDR = 40;

    __shared__ __align__(16) __bf16 sAh[128 * LDR];
    __shared__ __align__(16) __bf16 sAl[128 * LDR];
    __shared__ __align__(16) __bf16 sBh[128 * LDR];
    __shared__ __align__(16) __bf16 sBl[128 * LDR];

    const int tid  = threadIdx.x;
    const int lane = tid & 31;
    const int wid  = tid >> 5;
    const int wm   = wid & 3;
    const int wn   = wid >> 2;
    const int mBase = blockIdx.y * 128;
    const int nBase = blockIdx.x * 128;

    const v8f vzero = {0.f, 0.f, 0.f, 0.f, 0.f, 0.f, 0.f, 0.f};
    v8f acc[2][4];
#pragma unroll
    for (int i = 0; i < 2; ++i)
#pragma unroll
        for (int j = 0; j < 4; ++j) acc[i][j] = vzero;

    const int fm    = lane & 15;
    const int khalf = lane >> 4;
    const int crow  = tid >> 2;
    const int cc8   = (tid & 3) * 8;

    uint4 pf[2][4];

#define LOAD_CHUNK(K0)                                                        \
    _Pragma("unroll")                                                         \
    for (int rep = 0; rep < 2; ++rep) {                                       \
        const int row = rep * 64 + crow;                                      \
        const size_t gA = ((size_t)(mBase + row) << 12) + (K0) + cc8;         \
        const size_t gB = ((size_t)(nBase + row) << 12) + (K0) + cc8;         \
        pf[rep][0] = *(const uint4*)(XHI + gA);                               \
        pf[rep][1] = *(const uint4*)(XLO + gA);                               \
        pf[rep][2] = *(const uint4*)(WHI + gB);                               \
        pf[rep][3] = *(const uint4*)(WLO + gB);                               \
    }

    LOAD_CHUNK(0)

    for (int k0 = 0; k0 < KDIM; k0 += KC) {
#pragma unroll
        for (int rep = 0; rep < 2; ++rep) {
            const int lOff = (rep * 64 + crow) * LDR + cc8;
            *(uint4*)(sAh + lOff) = pf[rep][0];
            *(uint4*)(sAl + lOff) = pf[rep][1];
            *(uint4*)(sBh + lOff) = pf[rep][2];
            *(uint4*)(sBl + lOff) = pf[rep][3];
        }
        __syncthreads();

        if (k0 + KC < KDIM) { LOAD_CHUNK(k0 + KC) }

        v16bf ah[2], al[2], bh[4], bl[4];
#pragma unroll
        for (int mi = 0; mi < 2; ++mi) {
            const int m = wm * 32 + mi * 16 + fm;
            const __bf16* p = sAh + m * LDR + khalf * 8;
            const __bf16* q = sAl + m * LDR + khalf * 8;
            ah[mi] = pack16(p, p + 16);
            al[mi] = pack16(q, q + 16);
        }
#pragma unroll
        for (int ni = 0; ni < 4; ++ni) {
            const int n = wn * 64 + ni * 16 + fm;
            const __bf16* p = sBh + n * LDR + khalf * 8;
            const __bf16* q = sBl + n * LDR + khalf * 8;
            bh[ni] = pack16(p, p + 16);
            bl[ni] = pack16(q, q + 16);
        }

#pragma unroll
        for (int mi = 0; mi < 2; ++mi)
#pragma unroll
            for (int ni = 0; ni < 4; ++ni) {
                acc[mi][ni] = __builtin_amdgcn_wmma_f32_16x16x32_bf16(
                    false, ah[mi], false, bh[ni], (short)0, acc[mi][ni], false, false);
                acc[mi][ni] = __builtin_amdgcn_wmma_f32_16x16x32_bf16(
                    false, ah[mi], false, bl[ni], (short)0, acc[mi][ni], false, false);
                acc[mi][ni] = __builtin_amdgcn_wmma_f32_16x16x32_bf16(
                    false, al[mi], false, bh[ni], (short)0, acc[mi][ni], false, false);
            }
        __syncthreads();
    }
#undef LOAD_CHUNK

#pragma unroll
    for (int mi = 0; mi < 2; ++mi) {
#pragma unroll
        for (int ni = 0; ni < 4; ++ni) {
            const int m0 = mBase + wm * 32 + mi * 16 + khalf * 8;
            const int n  = nBase + wn * 64 + ni * 16 + fm;
#pragma unroll
            for (int r = 0; r < 8; ++r)
                Y[((size_t)(m0 + r) << 12) + n] = acc[mi][ni][r];
        }
    }
}
#endif  // USE_ASYNC_COPY

// ---------------------------------------------------------------------------
extern "C" void kernel_launch(void* const* d_in, const int* in_sizes, int n_in,
                              void* d_out, int out_size, void* d_ws, size_t ws_size,
                              hipStream_t stream) {
    (void)in_sizes; (void)n_in; (void)out_size; (void)ws_size;

    const float* x        = (const float*)d_in[0];
    const int*   Qidxs    = (const int*)  d_in[1];
    const float* SU       = (const float*)d_in[2];
    const float* SV       = (const float*)d_in[3];
    const float* Wscale   = (const float*)d_in[4];
    const float* codebook = (const float*)d_in[5];
    float* y = (float*)d_out;

    // Workspace layout (192 MB):
    //   [  0,  64MB) XHI : 8192x4096 bf16
    //   [ 64, 128MB) XLO : 8192x4096 bf16
    //   [128, 160MB) WHI : 4096x4096 bf16
    //   [160, 192MB) WLO : 4096x4096 bf16
    char* ws = (char*)d_ws;
    __bf16* XHI = (__bf16*)(ws);
    __bf16* XLO = (__bf16*)(ws + ((size_t)64  << 20));
    __bf16* WHI = (__bf16*)(ws + ((size_t)128 << 20));
    __bf16* WLO = (__bf16*)(ws + ((size_t)160 << 20));

    dequant_kernel<<<8192, 256, 0, stream>>>(Qidxs, codebook, WHI, WLO);
    fht_in_kernel<<<MROWS, 256, 0, stream>>>(x, SV, XHI, XLO);
    gemm_kernel<<<dim3(NOUT / 128, MROWS / 128), 256, 0, stream>>>(XHI, XLO, WHI, WLO, y);
    fht_out_kernel<<<MROWS, 256, 0, stream>>>(y, SU, Wscale);
}
